// LocalFeatureTransformer_47098611368207
// MI455X (gfx1250) — compile-verified, matrix operand order checked by reading, not run on hardware
//
#include <hip/hip_runtime.h>

// ---------------------------------------------------------------------------
// LocalFeatureTransformer for MI455X (gfx1250, wave32, WMMA bf16)
// ---------------------------------------------------------------------------
#define L_LEN 4800
#define C_DIM 256
#define C2_DIM 512
#define NHEAD 8
#define HDIM 32
#define TOPK 16
#define NLAYERS 8

typedef __attribute__((ext_vector_type(16))) __bf16 v16bf;
typedef __attribute__((ext_vector_type(8)))  float  v8f;

static __device__ __forceinline__ unsigned short f2bf(float f) {
  unsigned int u = __float_as_uint(f);
  unsigned int r = (u + 0x7FFFu + ((u >> 16) & 1u)) >> 16;
  return (unsigned short)r;
}

// Fragment element order (CDNA5 16x16x32 bf16): within a 32-wide k-tile,
// lane group khalf in {0,1}, element e in 0..15 maps to
//   k = 8*khalf + (e < 8 ? e : e + 8)
static __host__ __device__ __forceinline__ int frag_k(int q) {
  int khalf = q >> 4, e = q & 15;
  return khalf * 8 + (e < 8 ? e : e + 8);
}

// ---------------------------------------------------------------------------
// bf16 WMMA GEMM:  C[M,N] = A[M,K] @ B[K,N]
//   A: row-major bf16 (lda)      -> A frags = 2x b128 direct global loads
//   Bsw: pre-swizzled fragment-major bf16: [(k/32)*N + n]*32 + q
//        -> each B frag = 2x b128 contiguous global loads (L2-resident)
// Block: 256 threads = 8 waves; block tile 128x64; wave tile 16x64; K-step 32.
// No LDS, no barriers: steady state = 10 b128 loads + 4 wmma per K-step.
// ---------------------------------------------------------------------------
__global__ __launch_bounds__(256) void gemm_bf16(
    const unsigned short* __restrict__ A, int lda,
    const unsigned short* __restrict__ Bsw, int N,
    float* __restrict__ Cf, unsigned short* __restrict__ Cb, int ldc,
    int M, int Kd, int relu)
{
  const int tid = threadIdx.x;
  const int wid = tid >> 5, lane = tid & 31;
  const int rowBase = blockIdx.x * 128;
  const int colBase = blockIdx.y * 64;

  int m = rowBase + wid * 16 + (lane & 15);
  int mc = m < M ? m : (M - 1);                 // clamp: epilogue guards stores
  const int kb   = (lane >> 4) * 8;             // A-side k base within tile
  const int kh16 = (lane >> 4) * 16;            // B-side fragment sub-offset
  const int n0 = colBase + (lane & 15);

  const unsigned short* arow = A + (size_t)mc * lda;

  v8f acc0 = {}, acc1 = {}, acc2 = {}, acc3 = {};
  union F { v16bf v; uint4 q[2]; };

  for (int kt = 0; kt < Kd; kt += 32) {
    F af, b0, b1, b2, b3;
    // A fragment: two contiguous 16B chunks (k = kb..kb+7, kb+16..kb+23)
    af.q[0] = *(const uint4*)(arow + kt + kb);
    af.q[1] = *(const uint4*)(arow + kt + kb + 16);

    // B fragments: contiguous 32B each from the swizzled layout
    const unsigned short* bt = Bsw + ((size_t)(kt >> 5) * N) * 32;
    const uint4* p0 = (const uint4*)(bt + (size_t)(n0)      * 32 + kh16);
    const uint4* p1 = (const uint4*)(bt + (size_t)(n0 + 16) * 32 + kh16);
    const uint4* p2 = (const uint4*)(bt + (size_t)(n0 + 32) * 32 + kh16);
    const uint4* p3 = (const uint4*)(bt + (size_t)(n0 + 48) * 32 + kh16);
    b0.q[0] = p0[0]; b0.q[1] = p0[1];
    b1.q[0] = p1[0]; b1.q[1] = p1[1];
    b2.q[0] = p2[0]; b2.q[1] = p2[1];
    b3.q[0] = p3[0]; b3.q[1] = p3[1];

    if (kt + 32 < Kd) {  // pull next B k-tile toward L2/L0 (global_prefetch_b8)
      __builtin_prefetch(bt + ((size_t)N) * 32 + (size_t)n0 * 32 + kh16, 0, 1);
    }

    acc0 = __builtin_amdgcn_wmma_f32_16x16x32_bf16(false, af.v, false, b0.v,
                                                   (short)0, acc0, false, false);
    acc1 = __builtin_amdgcn_wmma_f32_16x16x32_bf16(false, af.v, false, b1.v,
                                                   (short)0, acc1, false, false);
    acc2 = __builtin_amdgcn_wmma_f32_16x16x32_bf16(false, af.v, false, b2.v,
                                                   (short)0, acc2, false, false);
    acc3 = __builtin_amdgcn_wmma_f32_16x16x32_bf16(false, af.v, false, b3.v,
                                                   (short)0, acc3, false, false);
  }

  // epilogue: C/D layout: VGPR r -> M = r (lanes 0-15) / M = 8+r (lanes 16-31)
  int mrow = rowBase + wid * 16 + ((lane >> 4) << 3);
  int c0 = colBase + (lane & 15);
#pragma unroll
  for (int r = 0; r < 8; ++r) {
    int row = mrow + r;
    if (row < M) {
      float v0 = acc0[r], v1 = acc1[r], v2 = acc2[r], v3 = acc3[r];
      if (relu) {
        v0 = fmaxf(v0, 0.f); v1 = fmaxf(v1, 0.f);
        v2 = fmaxf(v2, 0.f); v3 = fmaxf(v3, 0.f);
      }
      size_t p = (size_t)row * ldc + c0;
      if (Cf) { Cf[p] = v0; Cf[p + 16] = v1; Cf[p + 32] = v2; Cf[p + 48] = v3; }
      if (Cb) {
        Cb[p] = f2bf(v0); Cb[p + 16] = f2bf(v1);
        Cb[p + 32] = f2bf(v2); Cb[p + 48] = f2bf(v3);
      }
    }
  }
}

// ---------------------------------------------------------------------------
// swizzle f32 weights [NL][K][N] -> fragment-major bf16 [(NL)][(K/32)][N][32]
// ---------------------------------------------------------------------------
__global__ void swizzle_w(const float* __restrict__ W, unsigned short* __restrict__ out,
                          int Kd, int N, int total)
{
  int i = blockIdx.x * blockDim.x + threadIdx.x;
  if (i >= total) return;
  int per = Kd * N;
  int layer = i / per, rem = i % per;
  int q = rem & 31, n = (rem >> 5) % N, t = (rem >> 5) / N;
  int k = t * 32 + frag_k(q);
  out[i] = f2bf(W[(size_t)layer * per + (size_t)k * N + n]);
}

// swizzle a feature matrix used as B^T:  B[k][n] = F[n][k] * scale
// F: f32 [NROW][256]; out: [(256/32)][NROW][32]
__global__ void swizzle_featB(const float* __restrict__ F, unsigned short* __restrict__ out,
                              float scale, int nrow)
{
  int i = blockIdx.x * blockDim.x + threadIdx.x;
  int total = nrow * C_DIM;
  if (i >= total) return;
  int q = i & 31, n = (i >> 5) % nrow, t = (i >> 5) / nrow;
  int k = t * 32 + frag_k(q);
  out[i] = f2bf(F[(size_t)n * C_DIM + k] * scale);
}

__global__ __launch_bounds__(256) void init_features(
    const float* __restrict__ feat0, const float* __restrict__ feat1,
    float* __restrict__ X0, float* __restrict__ X1,
    unsigned short* __restrict__ f0b, unsigned short* __restrict__ f1b,
    unsigned short* __restrict__ f0s, unsigned short* __restrict__ f1s,
    unsigned short* __restrict__ Xcat0, unsigned short* __restrict__ Xcat1,
    float simScale)
{
  int i = blockIdx.x * blockDim.x + threadIdx.x;
  if (i >= L_LEN * C_DIM) return;
  int row = i >> 8, col = i & 255;
  float a = feat0[i], b = feat1[i];
  X0[i] = a; X1[i] = b;
  f0b[i] = f2bf(a); f1b[i] = f2bf(b);
  f0s[i] = f2bf(a * simScale); f1s[i] = f2bf(b * simScale);
  Xcat0[(size_t)row * C2_DIM + col] = f2bf(a);
  Xcat1[(size_t)row * C2_DIM + col] = f2bf(b);
}

// ---------------------------------------------------------------------------
// log-sum-exp stats over columns (reduce over rows) / over rows
// ---------------------------------------------------------------------------
__global__ void col_lse(const float* __restrict__ sim, const int* __restrict__ rmask,
                        float* __restrict__ cmax, float* __restrict__ clog,
                        int nr, int nc)
{
  int c = blockIdx.x * blockDim.x + threadIdx.x;
  if (c >= nc) return;
  float m = -3.0e38f, s = 0.f;
  for (int l = 0; l < nr; ++l) {
    float x = rmask[l] ? sim[(size_t)l * nc + c] : -1e30f;
    float nm = fmaxf(m, x);
    s = s * __expf(m - nm) + __expf(x - nm);
    m = nm;
  }
  cmax[c] = m;
  clog[c] = __logf(fmaxf(s, 1e-30f));
}

__global__ __launch_bounds__(256) void row_lse(
    const float* __restrict__ sim, const int* __restrict__ cmask,
    float* __restrict__ rmax, float* __restrict__ rlog, int nc)
{
  __shared__ float sm[256], ss[256];
  int row = blockIdx.x, tid = threadIdx.x;
  float m = -3.0e38f, s = 0.f;
  for (int c = tid; c < nc; c += 256) {
    float x = cmask[c] ? sim[(size_t)row * nc + c] : -1e30f;
    float nm = fmaxf(m, x);
    s = s * __expf(m - nm) + __expf(x - nm);
    m = nm;
  }
  sm[tid] = m; ss[tid] = s; __syncthreads();
  for (int off = 128; off > 0; off >>= 1) {
    if (tid < off) {
      float m2 = sm[tid + off], s2 = ss[tid + off];
      float nm = fmaxf(sm[tid], m2);
      ss[tid] = ss[tid] * __expf(sm[tid] - nm) + s2 * __expf(m2 - nm);
      sm[tid] = nm;
    }
    __syncthreads();
  }
  if (tid == 0) { rmax[row] = sm[0]; rlog[row] = __logf(fmaxf(ss[0], 1e-30f)); }
}

// ---------------------------------------------------------------------------
// top-16 by key = 2*sim - stat_max[e] - stat_log[e]  (monotonic in conf)
// ---------------------------------------------------------------------------
__global__ __launch_bounds__(256) void topk16(
    const float* __restrict__ sim, const float* __restrict__ smax,
    const float* __restrict__ slog, const int* __restrict__ emask,
    int* __restrict__ idxOut, int n, int strideFixed, int strideVar)
{
  __shared__ float sval[4096]; __shared__ int sidx[4096];
  __shared__ float rv[256];   __shared__ int ri[256];
  __shared__ int hp[256];     __shared__ int winner_s;
  int tid = threadIdx.x;
  size_t base = (size_t)blockIdx.x * strideFixed;

  float tv[16]; int tix[16];
#pragma unroll
  for (int q = 0; q < 16; ++q) { tv[q] = -3.0e38f; tix[q] = 0; }

  for (int e = tid; e < n; e += 256) {
    float key = emask[e]
        ? (2.f * sim[base + (size_t)e * strideVar] - smax[e] - slog[e])
        : -3.0e38f;
    int ix = e;
#pragma unroll
    for (int q = 0; q < 16; ++q) {
      if (key > tv[q]) {
        float t0 = tv[q]; int t1 = tix[q];
        tv[q] = key; tix[q] = ix; key = t0; ix = t1;
      }
    }
  }
#pragma unroll
  for (int q = 0; q < 16; ++q) { sval[tid * 16 + q] = tv[q]; sidx[tid * 16 + q] = tix[q]; }
  hp[tid] = 0;
  __syncthreads();

  for (int k = 0; k < TOPK; ++k) {
    rv[tid] = sval[tid * 16 + hp[tid]]; ri[tid] = tid;
    __syncthreads();
    for (int off = 128; off > 0; off >>= 1) {
      if (tid < off && rv[tid + off] > rv[tid]) { rv[tid] = rv[tid + off]; ri[tid] = ri[tid + off]; }
      __syncthreads();
    }
    if (tid == 0) {
      int w = ri[0]; winner_s = w;
      idxOut[(size_t)blockIdx.x * TOPK + k] = sidx[w * 16 + hp[w]];
    }
    __syncthreads();
    if (tid == winner_s) hp[tid]++;
    __syncthreads();
  }
}

// ---------------------------------------------------------------------------
// linear attention with Lq=1: msg = sum_j s_j v_j / (sum_j s_j + eps)
// ---------------------------------------------------------------------------
__global__ __launch_bounds__(256) void attn_kernel(
    const float* __restrict__ Q, const float* __restrict__ Kp,
    const float* __restrict__ Vp, const int* __restrict__ idx,
    const int* __restrict__ qmask, unsigned short* __restrict__ msg, int ntok)
{
  int wid = threadIdx.x >> 5, lane = threadIdx.x & 31;
  int tok = blockIdx.x * 8 + wid;
  if (tok >= ntok) return;

  int ids[TOPK];
#pragma unroll
  for (int j = 0; j < TOPK; ++j) ids[j] = idx[tok * TOPK + j];
  float mk = (float)qmask[tok];

#pragma unroll 1
  for (int h = 0; h < NHEAD; ++h) {
    int c = h * HDIM + lane;
    float q  = Q[(size_t)tok * C_DIM + c];
    float qp = (q > 0.f ? q + 1.f : __expf(q)) * mk;   // (elu(q)+1) * mask
    float denom = 0.f, acc = 0.f;
#pragma unroll 1
    for (int j = 0; j < TOPK; ++j) {
      int id = ids[j];
      float kv = Kp[(size_t)id * C_DIM + c];
      float kf = kv > 0.f ? kv + 1.f : __expf(kv);     // elu(k)+1
      float p = qp * kf;
#pragma unroll
      for (int o = 16; o > 0; o >>= 1) p += __shfl_xor(p, o, 32);  // head-dim dot
      denom += p;
      acc += p * Vp[(size_t)id * C_DIM + c];
    }
    msg[(size_t)tok * C_DIM + c] = f2bf(acc / (denom + 1e-6f));
  }
}

// ---------------------------------------------------------------------------
// layernorm -> bf16 into Xcat[:,256:512]
// ---------------------------------------------------------------------------
__global__ __launch_bounds__(256) void ln_to_cat(
    const float* __restrict__ Xin, const float* __restrict__ g,
    const float* __restrict__ b, unsigned short* __restrict__ Xcat)
{
  __shared__ float red[256]; __shared__ float bc[2];
  int row = blockIdx.x, tid = threadIdx.x;
  float v = Xin[(size_t)row * C_DIM + tid];
  red[tid] = v; __syncthreads();
  for (int off = 128; off > 0; off >>= 1) { if (tid < off) red[tid] += red[tid + off]; __syncthreads(); }
  if (tid == 0) bc[0] = red[0] * (1.f / C_DIM); __syncthreads();
  float d = v - bc[0];
  red[tid] = d * d; __syncthreads();
  for (int off = 128; off > 0; off >>= 1) { if (tid < off) red[tid] += red[tid + off]; __syncthreads(); }
  if (tid == 0) bc[1] = red[0] * (1.f / C_DIM); __syncthreads();
  float y = d * rsqrtf(bc[1] + 1e-5f) * g[tid] + b[tid];
  Xcat[(size_t)row * C2_DIM + C_DIM + tid] = f2bf(y);
}

// layernorm + residual; updates X (f32 master in d_out) and its bf16 shadow
__global__ __launch_bounds__(256) void ln_resid(
    const float* __restrict__ O, const float* __restrict__ g,
    const float* __restrict__ b, float* __restrict__ X,
    unsigned short* __restrict__ Xcat)
{
  __shared__ float red[256]; __shared__ float bc[2];
  int row = blockIdx.x, tid = threadIdx.x;
  float v = O[(size_t)row * C_DIM + tid];
  red[tid] = v; __syncthreads();
  for (int off = 128; off > 0; off >>= 1) { if (tid < off) red[tid] += red[tid + off]; __syncthreads(); }
  if (tid == 0) bc[0] = red[0] * (1.f / C_DIM); __syncthreads();
  float d = v - bc[0];
  red[tid] = d * d; __syncthreads();
  for (int off = 128; off > 0; off >>= 1) { if (tid < off) red[tid] += red[tid + off]; __syncthreads(); }
  if (tid == 0) bc[1] = red[0] * (1.f / C_DIM); __syncthreads();
  float y = d * rsqrtf(bc[1] + 1e-5f) * g[tid] + b[tid];
  size_t p = (size_t)row * C_DIM + tid;
  float xn = X[p] + y;
  X[p] = xn;
  Xcat[(size_t)row * C2_DIM + tid] = f2bf(xn);
}

// ---------------------------------------------------------------------------
// host orchestration
// ---------------------------------------------------------------------------
extern "C" void kernel_launch(void* const* d_in, const int* in_sizes, int n_in,
                              void* d_out, int out_size, void* d_ws, size_t ws_size,
                              hipStream_t stream) {
  (void)in_sizes; (void)n_in; (void)out_size; (void)ws_size;

  const float* feat0 = (const float*)d_in[0];
  const float* feat1 = (const float*)d_in[1];
  const float* Wq = (const float*)d_in[2];
  const float* Wk = (const float*)d_in[3];
  const float* Wv = (const float*)d_in[4];
  const float* Wm = (const float*)d_in[5];
  const float* W1 = (const float*)d_in[6];
  const float* W2 = (const float*)d_in[7];
  const float* g1 = (const float*)d_in[8];
  const float* b1 = (const float*)d_in[9];
  const float* g2 = (const float*)d_in[10];
  const float* b2 = (const float*)d_in[11];
  const int* mask0 = (const int*)d_in[12];
  const int* mask1 = (const int*)d_in[13];

  float* X0 = (float*)d_out;                       // f0 master lives in output
  float* X1 = X0 + (size_t)L_LEN * C_DIM;          // f1 master

  // ---- workspace carve-up ----
  char* ws = (char*)d_ws;
  size_t off = 0;
  auto alloc = [&](size_t bytes) -> char* {
    char* p = ws + off;
    off += (bytes + 255) & ~(size_t)255;
    return p;
  };
  const size_t FC = (size_t)L_LEN * C_DIM;
  unsigned short* f0b  = (unsigned short*)alloc(FC * 2);   // unscaled bf16 (A of K/V proj)
  unsigned short* f1b  = (unsigned short*)alloc(FC * 2);
  unsigned short* f0s  = (unsigned short*)alloc(FC * 2);   // scaled bf16 (A of sim)
  unsigned short* f1s  = (unsigned short*)alloc(FC * 2);
  unsigned short* f0sw = (unsigned short*)alloc(FC * 2);   // scaled, B-fragment swizzled
  unsigned short* f1sw = (unsigned short*)alloc(FC * 2);
  unsigned short* Wqsw = (unsigned short*)alloc((size_t)NLAYERS * C_DIM * C_DIM * 2);
  unsigned short* Wksw = (unsigned short*)alloc((size_t)NLAYERS * C_DIM * C_DIM * 2);
  unsigned short* Wvsw = (unsigned short*)alloc((size_t)NLAYERS * C_DIM * C_DIM * 2);
  unsigned short* Wmsw = (unsigned short*)alloc((size_t)NLAYERS * C_DIM * C_DIM * 2);
  unsigned short* W1sw = (unsigned short*)alloc((size_t)NLAYERS * C2_DIM * C2_DIM * 2);
  unsigned short* W2sw = (unsigned short*)alloc((size_t)NLAYERS * C2_DIM * C_DIM * 2);
  float* sim  = (float*)alloc((size_t)L_LEN * L_LEN * 4);
  float* cmax = (float*)alloc(L_LEN * 4);
  float* clog = (float*)alloc(L_LEN * 4);
  float* rmax = (float*)alloc(L_LEN * 4);
  float* rlog = (float*)alloc(L_LEN * 4);
  int* idx00 = (int*)alloc((size_t)L_LEN * TOPK * 4);
  int* idx01 = (int*)alloc((size_t)L_LEN * TOPK * 4);
  int* idx10 = (int*)alloc((size_t)L_LEN * TOPK * 4);
  int* idx11 = (int*)alloc((size_t)L_LEN * TOPK * 4);
  float* Qf = (float*)alloc(FC * 4);
  float* Kp = (float*)alloc(FC * 4);
  float* Vp = (float*)alloc(FC * 4);
  unsigned short* msgb = (unsigned short*)alloc(FC * 2);
  float* Mw = (float*)alloc(FC * 4);
  unsigned short* Hb = (unsigned short*)alloc((size_t)L_LEN * C2_DIM * 2);
  float* Of = (float*)alloc(FC * 4);
  unsigned short* Xcat0 = (unsigned short*)alloc((size_t)L_LEN * C2_DIM * 2);
  unsigned short* Xcat1 = (unsigned short*)alloc((size_t)L_LEN * C2_DIM * 2);

  // ---- one-time weight / feature-B swizzles (B operands are all static) ----
  auto swz = [&](const float* in, unsigned short* out, int Kd, int N) {
    int total = NLAYERS * Kd * N;
    swizzle_w<<<(total + 255) / 256, 256, 0, stream>>>(in, out, Kd, N, total);
  };
  swz(Wq, Wqsw, C_DIM, C_DIM);
  swz(Wk, Wksw, C_DIM, C_DIM);
  swz(Wv, Wvsw, C_DIM, C_DIM);
  swz(Wm, Wmsw, C_DIM, C_DIM);
  swz(W1, W1sw, C2_DIM, C2_DIM);
  swz(W2, W2sw, C2_DIM, C_DIM);

  // sim scale: (f0/sqrt(C))·(f1/sqrt(C))/T => each side scaled by 1/(16*sqrt(0.1))
  const float simScale = 0.1976423537f;
  init_features<<<(L_LEN * C_DIM + 255) / 256, 256, 0, stream>>>(
      feat0, feat1, X0, X1, f0b, f1b, f0s, f1s, Xcat0, Xcat1, simScale);
  swizzle_featB<<<(L_LEN * C_DIM + 255) / 256, 256, 0, stream>>>(feat0, f0sw, simScale, L_LEN);
  swizzle_featB<<<(L_LEN * C_DIM + 255) / 256, 256, 0, stream>>>(feat1, f1sw, simScale, L_LEN);

  auto gemm = [&](const unsigned short* A, int lda, const unsigned short* Bsw, int N,
                  float* Cf, unsigned short* Cb, int ldc, int M, int Kd, int relu) {
    dim3 grid((M + 127) / 128, N / 64);
    gemm_bf16<<<grid, 256, 0, stream>>>(A, lda, Bsw, N, Cf, Cb, ldc, M, Kd, relu);
  };

  // ---- conf_00: sim = f0s @ f0^T ----
  gemm(f0s, C_DIM, f0sw, L_LEN, sim, nullptr, L_LEN, L_LEN, C_DIM, 0);
  col_lse<<<(L_LEN + 255) / 256, 256, 0, stream>>>(sim, mask0, cmax, clog, L_LEN, L_LEN);
  topk16<<<L_LEN, 256, 0, stream>>>(sim, cmax, clog, mask0, idx00, L_LEN, L_LEN, 1);

  // ---- conf_01: sim = f0s @ f1^T ----
  gemm(f0s, C_DIM, f1sw, L_LEN, sim, nullptr, L_LEN, L_LEN, C_DIM, 0);
  col_lse<<<(L_LEN + 255) / 256, 256, 0, stream>>>(sim, mask0, cmax, clog, L_LEN, L_LEN);
  topk16<<<L_LEN, 256, 0, stream>>>(sim, cmax, clog, mask1, idx01, L_LEN, L_LEN, 1);
  row_lse<<<L_LEN, 256, 0, stream>>>(sim, mask1, rmax, rlog, L_LEN);
  topk16<<<L_LEN, 256, 0, stream>>>(sim, rmax, rlog, mask0, idx10, L_LEN, 1, L_LEN);

  // ---- conf_11: sim = f1s @ f1^T ----
  gemm(f1s, C_DIM, f1sw, L_LEN, sim, nullptr, L_LEN, L_LEN, C_DIM, 0);
  col_lse<<<(L_LEN + 255) / 256, 256, 0, stream>>>(sim, mask1, cmax, clog, L_LEN, L_LEN);
  topk16<<<L_LEN, 256, 0, stream>>>(sim, cmax, clog, mask1, idx11, L_LEN, L_LEN, 1);

  // ---- encoder layers ----
  const size_t WCC = (size_t)C_DIM * C_DIM;
  const size_t W1S = (size_t)C2_DIM * C2_DIM;
  const size_t W2S = (size_t)C2_DIM * C_DIM;

  for (int i = 0; i < NLAYERS; ++i) {
    const bool self = (i % 2) == 0;
    for (int s = 0; s < 2; ++s) {
      float* X = (s == 0) ? X0 : X1;
      unsigned short* Xcat = (s == 0) ? Xcat0 : Xcat1;
      const int* qm = (s == 0) ? mask0 : mask1;
      const unsigned short* srcb;
      const int* sidx;
      if (s == 0) { srcb = self ? f0b : f1b; sidx = self ? idx00 : idx01; }
      else        { srcb = self ? f1b : f0b; sidx = self ? idx11 : idx10; }

      // projections of the static source features — gathered later via idx
      gemm(srcb, C_DIM, Wksw + i * WCC, C_DIM, Kp, nullptr, C_DIM, L_LEN, C_DIM, 0);
      gemm(srcb, C_DIM, Wvsw + i * WCC, C_DIM, Vp, nullptr, C_DIM, L_LEN, C_DIM, 0);
      // Q from current X (bf16 shadow = first half of Xcat, lda = 512)
      gemm(Xcat, C2_DIM, Wqsw + i * WCC, C_DIM, Qf, nullptr, C_DIM, L_LEN, C_DIM, 0);

      attn_kernel<<<L_LEN / 8, 256, 0, stream>>>(Qf, Kp, Vp, sidx, qm, msgb, L_LEN);

      gemm(msgb, C_DIM, Wmsw + i * WCC, C_DIM, Mw, nullptr, C_DIM, L_LEN, C_DIM, 0);
      ln_to_cat<<<L_LEN, 256, 0, stream>>>(Mw, g1 + i * C_DIM, b1 + i * C_DIM, Xcat);

      // MLP: relu([X|msg] @ W1) @ W2
      gemm(Xcat, C2_DIM, W1sw + i * W1S, C2_DIM, nullptr, Hb, C2_DIM, L_LEN, C2_DIM, 1);
      gemm(Hb, C2_DIM, W2sw + i * W2S, C_DIM, Of, nullptr, C_DIM, L_LEN, C2_DIM, 0);

      ln_resid<<<L_LEN, 256, 0, stream>>>(Of, g2 + i * C_DIM, b2 + i * C_DIM, X, Xcat);
    }
  }
}
